// BahdanauAttentionDecoder_17248588661418
// MI455X (gfx1250) — matrix operand order repaired and checked by reading, not compiled, not run on hardware
//
#include <hip/hip_runtime.h>
#include <hip/hip_bf16.h>
#include <math.h>

// ---------------- problem constants (match reference) ----------------
#define Hh   1024
#define Ss   128
#define Bb   16
#define Tt   128
#define Vv   32000
#define BOS_TOK 1

typedef __attribute__((ext_vector_type(16))) __bf16 bf16x16;
typedef __attribute__((ext_vector_type(8)))  float  f32x8;

static __device__ __forceinline__ float sigf(float x) { return 1.0f / (1.0f + __expf(-x)); }

// WGP-scope prefetches (scope 0 pulls into all cache levels incl. WGP$).
// Two fixed-offset variants so both unrolled halves prefetch from the
// SAME un-incremented base register -> no extra v_add / hazard v_nops.
static __device__ __forceinline__ void prefetch_wgp_4k(const void* p) {
    asm volatile("global_prefetch_b8 %0, off offset:4096" :: "v"(p));
}
static __device__ __forceinline__ void prefetch_wgp_5k(const void* p) {
    asm volatile("global_prefetch_b8 %0, off offset:5120" :: "v"(p));
}

// ---------------------------------------------------------------------
// Pack fp32 weight W [N,K] (row-major) into WMMA-B bf16 tiles.
// Layout: [ntile][kblock(32)][lane(32)][j(16)]:
//   n = ntile*16 + (lane & 15)
//   k = kblock*32 + (lane < 16 ? 0 : 8) + (j < 8 ? j : j + 8)
// Each lane's 16 bf16 values contiguous -> two b128 loads per fragment.
// ---------------------------------------------------------------------
__global__ __launch_bounds__(256)
void pack_wmma_b(const float* __restrict__ W, __bf16* __restrict__ P,
                 int N, int K)
{
    long idx = (long)blockIdx.x * 256 + threadIdx.x;
    long total = (long)N * K;
    if (idx >= total) return;
    int  j    = (int)(idx & 15);
    long rest = idx >> 4;
    int  lane = (int)(rest & 31);
    rest >>= 5;
    int  nkb  = K >> 5;
    int  kb   = (int)(rest % nkb);
    long nt   = rest / nkb;
    int  n    = (int)(nt * 16) + (lane & 15);
    int  k    = kb * 32 + ((lane < 16) ? 0 : 8) + ((j < 8) ? j : j + 8);
    P[idx] = (__bf16)W[(long)n * K + k];
}

// ---------------------------------------------------------------------
// Pack fp32 activation A [M,K] (row-major, contiguous) into WMMA-A
// bf16 fragments. Layout: [mtile][kblock(32)][lane(32)][j(16)]:
//   m = mtile*16 + (lane & 15)
//   k = kblock*32 + (lane < 16 ? 0 : 8) + (j < 8 ? j : j + 8)
// (ISA 16-bit A layout: lanes 0-15 hold K {0..7,16..23}, lanes 16-31
//  hold K {8..15,24..31} of each 32-wide k-block.)
// ---------------------------------------------------------------------
__global__ __launch_bounds__(256)
void pack_wmma_a(const float* __restrict__ A, __bf16* __restrict__ P,
                 int K, long total)
{
    long idx = (long)blockIdx.x * 256 + threadIdx.x;
    if (idx >= total) return;
    int  j    = (int)(idx & 15);
    long rest = idx >> 4;
    int  lane = (int)(rest & 31);
    rest >>= 5;
    int  nkb  = K >> 5;
    int  kb   = (int)(rest % nkb);
    long mt   = rest / nkb;
    int  m    = (int)(mt * 16) + (lane & 15);
    int  k    = kb * 32 + ((lane < 16) ? 0 : 8) + ((j < 8) ? j : j + 8);
    P[idx] = (__bf16)A[(long)m * K + k];
}

// ---------------------------------------------------------------------
// Skinny GEMM: C = A @ B^T (+bias), A/B pre-packed bf16 fragments.
// Each wave owns TWO adjacent 16x16 output tiles (shared A fragment,
// two accumulator chains -> 2x matrix-pipe ILP, half the A traffic).
// K-loop manually unrolled by 2 (nkb is even: K = 1024 or 2048) so all
// loads/prefetches use immediate offsets from un-incremented bases.
// Steady state per 2 k-blocks: 4 WGP prefetches + 12 global_load_b128
// + 4 v_wmma_f32_16x16x32_bf16 + 3 pointer bumps.
// ---------------------------------------------------------------------
__global__ __launch_bounds__(256)
void gemm16x2_bf16_wmma(const __bf16* __restrict__ Ap,
                        const __bf16* __restrict__ Bp,
                        const float* __restrict__ bias,
                        float* __restrict__ C, int ldc,
                        int K, int Ntiles)      // Ntiles even, (K/32) even
{
    const int nkb   = K >> 5;
    const int lane  = threadIdx.x & 31;
    const int wave  = threadIdx.x >> 5;
    const int mtile = blockIdx.y;
    const int nt0   = (blockIdx.x * 8 + wave) * 2;
    if (nt0 >= Ntiles) return;

    const __bf16* aptr = Ap + (long)mtile * nkb * 512 + lane * 16;
    const __bf16* b0   = Bp + (long)nt0 * nkb * 512 + lane * 16;
    const __bf16* b1   = b0 + (long)nkb * 512;

    f32x8 acc0 = {0.f, 0.f, 0.f, 0.f, 0.f, 0.f, 0.f, 0.f};
    f32x8 acc1 = {0.f, 0.f, 0.f, 0.f, 0.f, 0.f, 0.f, 0.f};

    for (int kb = 0; kb < nkb; kb += 2) {
        // prefetch 4 k-blocks ahead for both B streams (WGP scope;
        // 32 lanes x 32B stride cover the whole 1KB tile's lines)
        prefetch_wgp_4k(b0);
        prefetch_wgp_5k(b0);
        prefetch_wgp_4k(b1);
        prefetch_wgp_5k(b1);

        bf16x16 af0  = *(const bf16x16*)(aptr);
        bf16x16 af1  = *(const bf16x16*)(aptr + 512);
        bf16x16 bf00 = *(const bf16x16*)(b0);
        bf16x16 bf01 = *(const bf16x16*)(b0 + 512);
        bf16x16 bf10 = *(const bf16x16*)(b1);
        bf16x16 bf11 = *(const bf16x16*)(b1 + 512);
        aptr += 1024; b0 += 1024; b1 += 1024;

        acc0 = __builtin_amdgcn_wmma_f32_16x16x32_bf16(
                   false, af0, false, bf00, (short)0, acc0, false, false);
        acc1 = __builtin_amdgcn_wmma_f32_16x16x32_bf16(
                   false, af0, false, bf10, (short)0, acc1, false, false);
        acc0 = __builtin_amdgcn_wmma_f32_16x16x32_bf16(
                   false, af1, false, bf01, (short)0, acc0, false, false);
        acc1 = __builtin_amdgcn_wmma_f32_16x16x32_bf16(
                   false, af1, false, bf11, (short)0, acc1, false, false);
    }

    const int  m     = lane & 15;
    const int  ccol0 = nt0 * 16 + m;
    const int  ccol1 = ccol0 + 16;
    const long crow0 = (long)mtile * 16 + ((lane < 16) ? 0 : 8);
    const float bi0  = bias ? bias[ccol0] : 0.0f;
    const float bi1  = bias ? bias[ccol1] : 0.0f;
#pragma unroll
    for (int r = 0; r < 8; ++r) {
        C[(crow0 + r) * (long)ldc + ccol0] = acc0[r] + bi0;
        C[(crow0 + r) * (long)ldc + ccol1] = acc1[r] + bi1;
    }
}

// ---------------------------------------------------------------------
// scores[b,s] = sum_h v_att[h] * tanh(q[b,h] + kproj[b,s,h])
// one wave per (b,s) pair
// ---------------------------------------------------------------------
__global__ __launch_bounds__(256)
void attn_scores(const float* __restrict__ q, const float* __restrict__ kproj,
                 const float* __restrict__ v_att, float* __restrict__ scores)
{
    int widx = blockIdx.x * 8 + (threadIdx.x >> 5);
    if (widx >= Bb * Ss) return;
    int lane = threadIdx.x & 31;
    int b = widx >> 7;                         // / Ss
    const float* qb = q + (long)b * Hh;
    const float* kp = kproj + (long)widx * Hh;
    float sum = 0.f;
    for (int h = lane; h < Hh; h += 32)
        sum += v_att[h] * tanhf(qb[h] + kp[h]);
#pragma unroll
    for (int off = 16; off; off >>= 1) sum += __shfl_xor(sum, off, 32);
    if (lane == 0) scores[widx] = sum;
}

// ---------------------------------------------------------------------
// Per-batch-row: softmax over S, write attw, ctx = w @ enc[b],
// build x = [emb[tok] | ctx]  (fp32, row stride 2H). One block per b.
// ---------------------------------------------------------------------
__global__ __launch_bounds__(256)
void softmax_ctx(const float* __restrict__ scores, const float* __restrict__ enc,
                 const float* __restrict__ emb, const int* __restrict__ tgt,
                 int t, float* __restrict__ x, float* __restrict__ attw)
{
    __shared__ float w[Ss];
    __shared__ float red[256];
    const int b = blockIdx.x, tid = threadIdx.x;

    float sv = (tid < Ss) ? scores[b * Ss + tid] : -INFINITY;
    red[tid] = sv; __syncthreads();
    for (int s2 = 128; s2 >= 1; s2 >>= 1) {
        if (tid < s2) red[tid] = fmaxf(red[tid], red[tid + s2]);
        __syncthreads();
    }
    const float mx = red[0]; __syncthreads();

    float ev = (tid < Ss) ? __expf(sv - mx) : 0.f;
    red[tid] = ev; __syncthreads();
    for (int s2 = 128; s2 >= 1; s2 >>= 1) {
        if (tid < s2) red[tid] += red[tid + s2];
        __syncthreads();
    }
    const float inv = 1.0f / red[0];
    if (tid < Ss) {
        float ww = ev * inv;
        w[tid] = ww;
        attw[((long)b * Tt + t) * Ss + tid] = ww;
    }
    __syncthreads();

    const int tok = (t == 0) ? BOS_TOK : tgt[b * Tt + (t - 1)];
    const float* erow = emb + (long)tok * Hh;
    for (int h = tid; h < Hh; h += 256) {
        x[(long)b * 2 * Hh + h] = erow[h];
        float acc = 0.f;
        const float* encb = enc + ((long)b * Ss) * Hh + h;
#pragma unroll 4
        for (int s = 0; s < Ss; ++s) acc += w[s] * encb[(long)s * Hh];
        x[(long)b * 2 * Hh + Hh + h] = acc;
    }
}

// ---------------------------------------------------------------------
// GRU gates (biases already added by the GEMMs, matching reference):
// r=sig(gx_r+gh_r); z=sig(gx_z+gh_z); n=tanh(gx_n + r*gh_n)
// h' = (1-z)*n + z*h
// ---------------------------------------------------------------------
__global__ __launch_bounds__(256)
void gru_elem(const float* __restrict__ gx, const float* __restrict__ gh,
              const float* __restrict__ hold, float* __restrict__ hnew)
{
    int i = blockIdx.x * 256 + threadIdx.x;      // over B*H
    if (i >= Bb * Hh) return;
    int b = i >> 10, h = i & (Hh - 1);
    const float* gxb = gx + (long)b * 3 * Hh;
    const float* ghb = gh + (long)b * 3 * Hh;
    float r = sigf(gxb[h] + ghb[h]);
    float z = sigf(gxb[Hh + h] + ghb[Hh + h]);
    float n = tanhf(gxb[2 * Hh + h] + r * ghb[2 * Hh + h]);
    hnew[i] = (1.0f - z) * n + z * hold[i];
}

// ---------------------------------------------------------------------
// log_softmax over V per batch row; writes out[b, t, :]
// ---------------------------------------------------------------------
__global__ __launch_bounds__(256)
void log_softmax_row(const float* __restrict__ logits, float* __restrict__ out, int t)
{
    __shared__ float red[256];
    const int b = blockIdx.x, tid = threadIdx.x;
    const float* row = logits + (long)b * Vv;

    float mx = -INFINITY;
    for (int v = tid; v < Vv; v += 256) mx = fmaxf(mx, row[v]);
    red[tid] = mx; __syncthreads();
    for (int s2 = 128; s2 >= 1; s2 >>= 1) {
        if (tid < s2) red[tid] = fmaxf(red[tid], red[tid + s2]);
        __syncthreads();
    }
    mx = red[0]; __syncthreads();

    float sum = 0.f;
    for (int v = tid; v < Vv; v += 256) sum += __expf(row[v] - mx);
    red[tid] = sum; __syncthreads();
    for (int s2 = 128; s2 >= 1; s2 >>= 1) {
        if (tid < s2) red[tid] += red[tid + s2];
        __syncthreads();
    }
    const float lse = mx + __logf(red[0]);
    float* orow = out + ((long)b * Tt + t) * (long)Vv;
    for (int v = tid; v < Vv; v += 256) orow[v] = row[v] - lse;
}

// ---------------------------------------------------------------------
extern "C" void kernel_launch(void* const* d_in, const int* in_sizes, int n_in,
                              void* d_out, int out_size, void* d_ws, size_t ws_size,
                              hipStream_t stream)
{
    const float* enc   = (const float*)d_in[0];   // [B,S,H]
    const float* ehid  = (const float*)d_in[1];   // [1,B,H]
    const int*   tgt   = (const int*)  d_in[2];   // [B,T]
    const float* emb   = (const float*)d_in[3];   // [V,H]
    const float* Wq    = (const float*)d_in[4];   // [H,H]
    const float* Wk    = (const float*)d_in[5];   // [H,H]
    const float* v_att = (const float*)d_in[6];   // [H]
    const float* W_ih  = (const float*)d_in[7];   // [3H,2H]
    const float* W_hh  = (const float*)d_in[8];   // [3H,H]
    const float* b_ih  = (const float*)d_in[9];   // [3H]
    const float* b_hh  = (const float*)d_in[10];  // [3H]
    const float* W_out = (const float*)d_in[11];  // [V,H]
    const float* b_out = (const float*)d_in[12];  // [V]
    float* out = (float*)d_out;

    // ---- workspace carve-up (256B aligned) ----
    char* ws = (char*)d_ws;
    size_t off = 0;
    auto carve = [&](size_t bytes) -> char* {
        char* p = ws + off;
        off = (off + bytes + 255) & ~(size_t)255;
        return p;
    };
    __bf16* pWout = (__bf16*)carve((size_t)Vv * Hh * 2);
    __bf16* pWih  = (__bf16*)carve((size_t)3 * Hh * 2 * Hh * 2);
    __bf16* pWhh  = (__bf16*)carve((size_t)3 * Hh * Hh * 2);
    __bf16* pWq   = (__bf16*)carve((size_t)Hh * Hh * 2);
    __bf16* pWk   = (__bf16*)carve((size_t)Hh * Hh * 2);
    __bf16* pEnc  = (__bf16*)carve((size_t)Bb * Ss * Hh * 2);  // packed A for kproj
    __bf16* phI   = (__bf16*)carve((size_t)Bb * Hh * 2);       // packed h (init)
    __bf16* phA   = (__bf16*)carve((size_t)Bb * Hh * 2);
    __bf16* phB   = (__bf16*)carve((size_t)Bb * Hh * 2);
    __bf16* pX    = (__bf16*)carve((size_t)Bb * 2 * Hh * 2);   // packed x
    float* kproj  = (float*)carve((size_t)Bb * Ss * Hh * 4);
    float* qbuf   = (float*)carve((size_t)Bb * Hh * 4);
    float* scores = (float*)carve((size_t)Bb * Ss * 4);
    float* xbuf   = (float*)carve((size_t)Bb * 2 * Hh * 4);
    float* gx     = (float*)carve((size_t)Bb * 3 * Hh * 4);
    float* gh     = (float*)carve((size_t)Bb * 3 * Hh * 4);
    float* hA     = (float*)carve((size_t)Bb * Hh * 4);
    float* hB     = (float*)carve((size_t)Bb * Hh * 4);
    float* logits = (float*)carve((size_t)Bb * Vv * 4);

    const size_t LOGP_ELEMS = (size_t)Bb * Tt * Vv;
    const size_t HT_ELEMS   = (size_t)Bb * Hh;
    float* attw_out = out + LOGP_ELEMS + HT_ELEMS;

    // ---- prep: bf16 weight + activation packing (L2-resident set) ----
    auto packB = [&](const float* Wsrc, __bf16* dst, int N, int K) {
        long total = (long)N * K;
        pack_wmma_b<<<(int)((total + 255) / 256), 256, 0, stream>>>(Wsrc, dst, N, K);
    };
    auto packA = [&](const float* Asrc, __bf16* dst, int M, int K) {
        long total = (long)M * K;
        pack_wmma_a<<<(int)((total + 255) / 256), 256, 0, stream>>>(Asrc, dst, K, total);
    };
    packB(W_out, pWout, Vv, Hh);
    packB(W_ih,  pWih,  3 * Hh, 2 * Hh);
    packB(W_hh,  pWhh,  3 * Hh, Hh);
    packB(Wq,    pWq,   Hh, Hh);
    packB(Wk,    pWk,   Hh, Hh);
    packA(enc,  pEnc, Bb * Ss, Hh);
    packA(ehid, phI,  Bb, Hh);

    // kproj = enc @ Wk^T : M=2048 (128 row-tiles), N=1024 (64 tiles -> 32 pairs)
    gemm16x2_bf16_wmma<<<dim3(4, (Bb * Ss) / 16), 256, 0, stream>>>(
        pEnc, pWk, nullptr, kproj, Hh, Hh, Hh / 16);

    // ---- sequential decode: 128 steps ----
    const float*  hcur  = ehid;
    const __bf16* phcur = phI;
    float*  hnext  = hA;
    __bf16* phnext = phA;
    for (int t = 0; t < Tt; ++t) {
        // q = h @ Wq^T   (64 tiles -> 32 pairs -> 4 blocks)
        gemm16x2_bf16_wmma<<<dim3(4, 1), 256, 0, stream>>>(
            phcur, pWq, nullptr, qbuf, Hh, Hh, Hh / 16);
        // additive attention scores
        attn_scores<<<(Bb * Ss) / 8, 256, 0, stream>>>(qbuf, kproj, v_att, scores);
        // softmax + attw output + ctx + x = [emb | ctx]
        softmax_ctx<<<Bb, 256, 0, stream>>>(scores, enc, emb, tgt, t, xbuf, attw_out);
        // pack x for WMMA-A
        packA(xbuf, pX, Bb, 2 * Hh);
        // gx = x @ W_ih^T + b_ih   (192 tiles -> 96 pairs -> 12 blocks, K=2048)
        gemm16x2_bf16_wmma<<<dim3(12, 1), 256, 0, stream>>>(
            pX, pWih, b_ih, gx, 3 * Hh, 2 * Hh, (3 * Hh) / 16);
        // gh = h @ W_hh^T + b_hh
        gemm16x2_bf16_wmma<<<dim3(12, 1), 256, 0, stream>>>(
            phcur, pWhh, b_hh, gh, 3 * Hh, Hh, (3 * Hh) / 16);
        // GRU update + pack h' for WMMA-A
        gru_elem<<<(Bb * Hh) / 256, 256, 0, stream>>>(gx, gh, hcur, hnext);
        packA(hnext, phnext, Bb, Hh);
        // logits = h' @ W_out^T + b_out  (2000 tiles -> 1000 pairs -> 125 blocks)
        gemm16x2_bf16_wmma<<<dim3(125, 1), 256, 0, stream>>>(
            phnext, pWout, b_out, logits, Vv, Hh, Vv / 16);
        // log_softmax -> out[b, t, :]
        log_softmax_row<<<Bb, 256, 0, stream>>>(logits, out, t);

        hcur  = hnext;  phcur = phnext;
        hnext  = (hcur == hA) ? hB : hA;
        phnext = (phcur == phA) ? phB : phA;
    }

    // final hidden state -> out[B*T*V ..]
    hipMemcpyAsync(out + LOGP_ELEMS, hcur, HT_ELEMS * sizeof(float),
                   hipMemcpyDeviceToDevice, stream);
}
